// GraphSAGEEnrichedRecommender_35210141893412
// MI455X (gfx1250) — compile-verified, matrix operand order checked by reading, not run on hardware
//
#include <hip/hip_runtime.h>

// ---------------------------------------------------------------------------
// GraphSAGE (mean / max / mean) + MLP head for gfx1250 (MI455X, wave32, WMMA)
// GEMMs: bf16 WMMA (16x16x32, f32 accumulate), A staged in LDS per workgroup.
// ---------------------------------------------------------------------------

typedef __attribute__((ext_vector_type(16))) __bf16 v16bf;
typedef __attribute__((ext_vector_type(8)))  float  v8f;

static __device__ __forceinline__ unsigned short bf16_bits(float f)
{
    __bf16 h = (__bf16)f;
    return *(const unsigned short*)&h;
}

// ---------------------------------------------------------------------------
// Weight repack: fp32 [K,F] row-major -> per-lane bf16 WMMA B-fragments.
// B is 32x16 (KxN) per 16x16x32 step. Per ISA 16-bit operand layout:
//   lane L holds column N = L&15;  hi = L>>4
//   element e<8 : K = kt*32 + 8*hi + e
//   element e>=8: K = kt*32 + 8*hi + 16 + (e-8)
// Fragment for (nt,kt,lane) stored contiguously: 16 bf16 = 32 bytes.
// ---------------------------------------------------------------------------
__global__ void pack_w(const float* __restrict__ W, unsigned short* __restrict__ Wp,
                       int K, int F, int Fpad)
{
    const int nkt   = K >> 5;
    const int total = (Fpad >> 4) * nkt * 32;
    int t = blockIdx.x * blockDim.x + threadIdx.x;
    if (t >= total) return;
    const int lane = t & 31;
    const int tile = t >> 5;
    const int nt   = tile / nkt;
    const int kt   = tile % nkt;
    const int n    = (nt << 4) + (lane & 15);
    const int kb   = (kt << 5) + ((lane >> 4) << 3);
    unsigned short* dst = Wp + ((size_t)t << 4);
#pragma unroll
    for (int e = 0; e < 16; ++e) {
        const int k = kb + (e < 8 ? e : e + 8);
        dst[e] = bf16_bits((n < F) ? W[(size_t)k * F + n] : 0.0f);
    }
}

// ---------------------------------------------------------------------------
// Aggregation kernels (HBM/atomic bound: ~2.5 GB edge traffic total)
// ---------------------------------------------------------------------------
__global__ void count_deg(const int* __restrict__ dst, int* __restrict__ cnt, int E)
{
    int e = blockIdx.x * blockDim.x + threadIdx.x;
    if (e < E) atomicAdd(&cnt[dst[e]], 1);
}

__global__ void scatter_add4(const float* __restrict__ x, const int* __restrict__ src,
                             const int* __restrict__ dst, float* __restrict__ aggr,
                             int E, int d)
{
    const int per = d >> 2;
    long t = (long)blockIdx.x * blockDim.x + threadIdx.x;
    if (t >= (long)E * per) return;
    const int e = (int)(t / per);
    const int c = (int)(t % per) << 2;
    const float4 v = *(const float4*)(x + (size_t)src[e] * d + c);
    float* p = aggr + (size_t)dst[e] * d + c;
    atomicAdd(p + 0, v.x); atomicAdd(p + 1, v.y);
    atomicAdd(p + 2, v.z); atomicAdd(p + 3, v.w);
}

// max aggregation: inputs are post-ReLU (>=0) so uint-bit atomicMax is exact;
// zero init == reference's "empty segment -> 0" semantics.
__global__ void scatter_max4(const float* __restrict__ x, const int* __restrict__ src,
                             const int* __restrict__ dst, float* __restrict__ aggr,
                             int E, int d)
{
    const int per = d >> 2;
    long t = (long)blockIdx.x * blockDim.x + threadIdx.x;
    if (t >= (long)E * per) return;
    const int e = (int)(t / per);
    const int c = (int)(t % per) << 2;
    const float4 v = *(const float4*)(x + (size_t)src[e] * d + c);
    unsigned int* p = (unsigned int*)(aggr + (size_t)dst[e] * d + c);
    atomicMax(p + 0, __float_as_uint(fmaxf(v.x, 0.0f)));
    atomicMax(p + 1, __float_as_uint(fmaxf(v.y, 0.0f)));
    atomicMax(p + 2, __float_as_uint(fmaxf(v.z, 0.0f)));
    atomicMax(p + 3, __float_as_uint(fmaxf(v.w, 0.0f)));
}

__global__ void mean_div4(float* __restrict__ aggr, const int* __restrict__ cnt,
                          int N, int d)
{
    const int per = d >> 2;
    long t = (long)blockIdx.x * blockDim.x + threadIdx.x;
    if (t >= (long)N * per) return;
    const int n = (int)(t / per);
    const int c = (int)(t % per) << 2;
    const float inv = 1.0f / fmaxf((float)cnt[n], 1.0f);
    float4* p = (float4*)(aggr + (size_t)n * d + c);
    float4 v = *p;
    v.x *= inv; v.y *= inv; v.z *= inv; v.w *= inv;
    *p = v;
}

// ---------------------------------------------------------------------------
// Fused WMMA GEMM: out = [LN+ReLU]( A1@W1 + A2@W2 + bias )
//   - block = 256 threads = 8 wave32s; block owns 16 rows x Fpad cols
//   - phase 1: cooperative fp32->bf16 convert of the 16xK A tiles into LDS
//              (once per BLOCK, not per wave: 8x less global A traffic)
//   - phase 2: kt loop with running pointers; A fragment from LDS
//              (2x ds_load_b128), B fragment from packed weights
//              (2x global_load_b128, L2-resident).
//              TILES template param = N-tiles per wave (compile-time), so the
//              2-tile variant issues two independent back-to-back WMMAs per A
//              fragment with NO exec-mask guard -> hides WMMA->WMMA hazard.
//   - phase 3: bias + LayerNorm + ReLU fused via LDS + half-wave shuffles
// A fragment per ISA 16-bit layout: lane L: row M=L&15, hi=L>>4;
//   elems 0..7 -> K = kb..kb+7 (kb = kt*32 + 8*hi), elems 8..15 -> kb+16..kb+23
// ---------------------------------------------------------------------------
#define AL_STRIDE (16 * 256)   // shorts per A-stage region (max K = 256)

template<int TILES>
__global__ __launch_bounds__(256)
void sage_gemm_ln(const float* __restrict__ A1, const unsigned short* __restrict__ W1,
                  const float* __restrict__ A2, const unsigned short* __restrict__ W2,
                  const float* __restrict__ bias, const float* __restrict__ lng,
                  const float* __restrict__ lnb, float* __restrict__ out,
                  int K1, int K2, int Fpad, int Fout, int do_ln)
{
    __shared__ unsigned short aLds[2 * AL_STRIDE];  // 16 KB: bf16 A tiles
    __shared__ float          cLds[16 * 256];       // 16 KB: C block for LN

    const int tid  = threadIdx.x;
    const int lane = tid & 31;
    const int wave = tid >> 5;
    const int hi   = lane >> 4;
    const int lm   = lane & 15;
    const int row0 = blockIdx.x * 16;
    const int ntN  = Fpad >> 4;

    // ---- phase 1: stage A tiles into LDS as bf16 (row-major [16][K]) ----
    {
        const float* Ap = A1;
        int          K  = K1;
#pragma unroll
        for (int p = 0; p < 2; ++p) {
            if (Ap != nullptr) {
                unsigned short* dst  = aLds + p * AL_STRIDE;
                const int kdiv = K >> 2;             // float4s per row (pow2)
                const int sh   = __builtin_ctz(kdiv);
                const int msk  = kdiv - 1;
                for (int idx = tid; idx < (16 * K) >> 2; idx += 256) {
                    const int row = idx >> sh;
                    const int col = (idx & msk) << 2;
                    const float4 v = *(const float4*)(Ap + (size_t)(row0 + row) * K + col);
                    ushort4 h;
                    h.x = bf16_bits(v.x); h.y = bf16_bits(v.y);
                    h.z = bf16_bits(v.z); h.w = bf16_bits(v.w);
                    *(ushort4*)(dst + row * K + col) = h;
                }
            }
            Ap = A2; K = K2;
        }
    }
    __syncthreads();

    // ---- phase 2: WMMA main loops ----
    const int  nt0  = wave;
    const int  nt1  = wave + 8;
    const bool has0 = nt0 < ntN;

    v8f acc0 = {0.f, 0.f, 0.f, 0.f, 0.f, 0.f, 0.f, 0.f};
    v8f acc1 = {0.f, 0.f, 0.f, 0.f, 0.f, 0.f, 0.f, 0.f};

    if (has0) {
        const unsigned short* Wptr = W1;
        int                   K    = K1;
        bool                  live = (A1 != nullptr);
#pragma unroll
        for (int p = 0; p < 2; ++p) {
            if (live) {
                const int nkt = K >> 5;
                // running pointers: A advances 32 shorts (64 B) per kt;
                // B advances one fragment-row = 32 lanes * 32 B = 1024 B per kt
                const unsigned short* ap = aLds + p * AL_STRIDE + lm * K + (hi << 3);
                const uint4* wp0 = (const uint4*)Wptr + ((size_t)nt0 * nkt * 32 + lane) * 2;
                const uint4* wp1 = (const uint4*)Wptr + ((size_t)nt1 * nkt * 32 + lane) * 2;
                for (int kt = 0; kt < nkt; ++kt) {
                    union { uint4 q[2]; v16bf v; } aa;
                    aa.q[0] = *(const uint4*)(ap);        // K = kb..kb+7
                    aa.q[1] = *(const uint4*)(ap + 16);   // K = kb+16..kb+23

                    union { uint4 q[2]; v16bf v; } b0;
                    b0.q[0] = wp0[0]; b0.q[1] = wp0[1];
                    acc0 = __builtin_amdgcn_wmma_f32_16x16x32_bf16(
                        false, aa.v, false, b0.v, (short)0, acc0, false, false);

                    if (TILES == 2) {
                        union { uint4 q[2]; v16bf v; } b1;
                        b1.q[0] = wp1[0]; b1.q[1] = wp1[1];
                        acc1 = __builtin_amdgcn_wmma_f32_16x16x32_bf16(
                            false, aa.v, false, b1.v, (short)0, acc1, false, false);
                    }
                    ap  += 32;
                    wp0 += 64;
                    wp1 += 64;
                }
            }
            Wptr = W2; K = K2; live = (A2 != nullptr);
        }
    }

    // ---- phase 3: epilogue ----
    // C/D layout: VGPR r -> row M = r + 8*hi, col N = lane&15
    if (do_ln) {
        if (has0) {
            const int n0 = (nt0 << 4) + lm;
            const float bn0 = bias[n0];
#pragma unroll
            for (int r = 0; r < 8; ++r)
                cLds[(8 * hi + r) * Fpad + n0] = acc0[r] + bn0;
            if (TILES == 2) {
                const int n1 = (nt1 << 4) + lm;
                const float bn1 = bias[n1];
#pragma unroll
                for (int r = 0; r < 8; ++r)
                    cLds[(8 * hi + r) * Fpad + n1] = acc1[r] + bn1;
            }
        }
        __syncthreads();
        // 16 row-teams of 16 threads (each team lives inside one half-wave)
        const int r  = tid >> 4;
        const int c0 = tid & 15;
        float s = 0.0f, ss = 0.0f;
        for (int c = c0; c < Fpad; c += 16) {
            const float v = cLds[r * Fpad + c];
            s += v; ss += v * v;
        }
#pragma unroll
        for (int off = 8; off > 0; off >>= 1) {
            s  += __shfl_xor(s,  off, 16);
            ss += __shfl_xor(ss, off, 16);
        }
        const float inv  = 1.0f / (float)Fpad;
        const float mean = s * inv;
        float var = ss * inv - mean * mean;
        var = fmaxf(var, 0.0f);
        const float rstd = rsqrtf(var + 1e-5f);
        for (int c = c0; c < Fpad; c += 16) {
            const float v = (cLds[r * Fpad + c] - mean) * rstd * lng[c] + lnb[c];
            out[(size_t)(row0 + r) * Fpad + c] = fmaxf(v, 0.0f);
        }
    } else {
        if (has0) {
            const int n0 = (nt0 << 4) + lm;
            if (n0 < Fout) {
                const float bn0 = bias[n0];
#pragma unroll
                for (int r = 0; r < 8; ++r)
                    out[(size_t)(row0 + 8 * hi + r) * Fout + n0] = acc0[r] + bn0;
            }
            if (TILES == 2) {
                const int n1 = (nt1 << 4) + lm;
                if (n1 < Fout) {
                    const float bn1 = bias[n1];
#pragma unroll
                    for (int r = 0; r < 8; ++r)
                        out[(size_t)(row0 + 8 * hi + r) * Fout + n1] = acc1[r] + bn1;
                }
            }
        }
    }
}

// ---------------------------------------------------------------------------
extern "C" void kernel_launch(void* const* d_in, const int* in_sizes, int n_in,
                              void* d_out, int out_size, void* d_ws, size_t ws_size,
                              hipStream_t stream)
{
    const float* x     = (const float*)d_in[0];
    const int*   src   = (const int*)  d_in[1];
    const int*   dst   = (const int*)  d_in[2];
    const float* s1_wl = (const float*)d_in[3];
    const float* s1_bl = (const float*)d_in[4];
    const float* s1_wr = (const float*)d_in[5];
    const float* ln1_g = (const float*)d_in[6];
    const float* ln1_b = (const float*)d_in[7];
    const float* s2_wl = (const float*)d_in[8];
    const float* s2_bl = (const float*)d_in[9];
    const float* s2_wr = (const float*)d_in[10];
    const float* ln2_g = (const float*)d_in[11];
    const float* ln2_b = (const float*)d_in[12];
    const float* s3_wl = (const float*)d_in[13];
    const float* s3_bl = (const float*)d_in[14];
    const float* s3_wr = (const float*)d_in[15];
    const float* ln3_g = (const float*)d_in[16];
    const float* ln3_b = (const float*)d_in[17];
    const float* l1_w  = (const float*)d_in[18];
    const float* l1_b  = (const float*)d_in[19];
    const float* ln4_g = (const float*)d_in[20];
    const float* ln4_b = (const float*)d_in[21];
    const float* lo_w  = (const float*)d_in[22];
    const float* lo_b  = (const float*)d_in[23];
    float* out = (float*)d_out;

    const int N = 100000, E = 800000;

    // ---- workspace layout (aliased buffers, ~257 MB) ----
    char*  ws  = (char*)d_ws;
    size_t off = 0;
    auto walloc = [&](size_t bytes) -> void* {
        void* p = ws + off;
        off += (bytes + 255) & ~(size_t)255;
        return p;
    };
    int*   cnt  = (int*)  walloc((size_t)N * 4);
    float* bufA = (float*)walloc((size_t)N * 256 * 4);  // aggr1 -> aggr2 -> aggr3 -> h4
    float* bufB = (float*)walloc((size_t)N * 256 * 4);  // h2
    float* bufH = (float*)walloc((size_t)N * 128 * 4);  // h1 -> h3
    unsigned short* p_s1wl = (unsigned short*)walloc(128 * 128 * 2);
    unsigned short* p_s1wr = (unsigned short*)walloc(128 * 128 * 2);
    unsigned short* p_s2wl = (unsigned short*)walloc(128 * 256 * 2);
    unsigned short* p_s2wr = (unsigned short*)walloc(128 * 256 * 2);
    unsigned short* p_s3wl = (unsigned short*)walloc(256 * 128 * 2);
    unsigned short* p_s3wr = (unsigned short*)walloc(256 * 128 * 2);
    unsigned short* p_l1   = (unsigned short*)walloc(128 *  64 * 2);
    unsigned short* p_lo   = (unsigned short*)walloc( 64 * 112 * 2);

    auto cdiv = [](long a, long b) -> int { return (int)((a + b - 1) / b); };

    // ---- pack weights to WMMA bf16 fragment layout ----
    auto pack = [&](const float* W, unsigned short* Wp, int K, int F, int Fpad) {
        const int total = (Fpad >> 4) * (K >> 5) * 32;
        pack_w<<<cdiv(total, 256), 256, 0, stream>>>(W, Wp, K, F, Fpad);
    };
    pack(s1_wl, p_s1wl, 128, 128, 128);
    pack(s1_wr, p_s1wr, 128, 128, 128);
    pack(s2_wl, p_s2wl, 128, 256, 256);
    pack(s2_wr, p_s2wr, 128, 256, 256);
    pack(s3_wl, p_s3wl, 256, 128, 128);
    pack(s3_wr, p_s3wr, 256, 128, 128);
    pack(l1_w,  p_l1,   128,  64,  64);
    pack(lo_w,  p_lo,    64, 100, 112);

    const int rowBlocks = N / 16;  // 6250

    // ---- degree counts (shared by both mean layers) ----
    hipMemsetAsync(cnt, 0, (size_t)N * 4, stream);
    count_deg<<<cdiv(E, 256), 256, 0, stream>>>(dst, cnt, E);

    // ---- layer 1: SAGE mean, 128 -> 128 ----
    hipMemsetAsync(bufA, 0, (size_t)N * 128 * 4, stream);
    scatter_add4<<<cdiv((long)E * 32, 256), 256, 0, stream>>>(x, src, dst, bufA, E, 128);
    mean_div4<<<cdiv((long)N * 32, 256), 256, 0, stream>>>(bufA, cnt, N, 128);
    sage_gemm_ln<1><<<rowBlocks, 256, 0, stream>>>(
        bufA, p_s1wl, x, p_s1wr, s1_bl, ln1_g, ln1_b, bufH,
        128, 128, 128, 128, 1);                                   // h1 = bufH

    // ---- layer 2: SAGE max, 128 -> 256 ----
    hipMemsetAsync(bufA, 0, (size_t)N * 128 * 4, stream);         // +0.0 = identity (h1 >= 0)
    scatter_max4<<<cdiv((long)E * 32, 256), 256, 0, stream>>>(bufH, src, dst, bufA, E, 128);
    sage_gemm_ln<2><<<rowBlocks, 256, 0, stream>>>(
        bufA, p_s2wl, bufH, p_s2wr, s2_bl, ln2_g, ln2_b, bufB,
        128, 128, 256, 256, 1);                                   // h2 = bufB

    // ---- layer 3: SAGE mean, 256 -> 128 ----
    hipMemsetAsync(bufA, 0, (size_t)N * 256 * 4, stream);
    scatter_add4<<<cdiv((long)E * 64, 256), 256, 0, stream>>>(bufB, src, dst, bufA, E, 256);
    mean_div4<<<cdiv((long)N * 64, 256), 256, 0, stream>>>(bufA, cnt, N, 256);
    sage_gemm_ln<1><<<rowBlocks, 256, 0, stream>>>(
        bufA, p_s3wl, bufB, p_s3wr, s3_bl, ln3_g, ln3_b, bufH,
        256, 256, 128, 128, 1);                                   // h3 = bufH

    // ---- MLP head: 128 -> 64 (LN+ReLU) -> 100 ----
    sage_gemm_ln<1><<<rowBlocks, 256, 0, stream>>>(
        bufH, p_l1, nullptr, nullptr, l1_b, ln4_g, ln4_b, bufA,
        128, 0, 64, 64, 1);                                       // h4 = bufA (stride 64)
    sage_gemm_ln<1><<<rowBlocks, 256, 0, stream>>>(
        bufA, p_lo, nullptr, nullptr, lo_b, nullptr, nullptr, out,
        64, 0, 112, 100, 0);                                      // logits, no LN
}
